// LipFormerDecoder_2903397892830
// MI455X (gfx1250) — matrix operand order repaired and checked by reading, not compiled, not run on hardware
//
#include <hip/hip_runtime.h>
#include <hip/hip_bf16.h>

typedef __attribute__((ext_vector_type(16))) __bf16 v16bf;
typedef __attribute__((ext_vector_type(8)))  __bf16 v8bf;
typedef __attribute__((ext_vector_type(8)))  float  v8f;

#define BB   64
#define TT   100
#define DD   1024
#define HH   1024
#define H3   3072
#define OUTD 28

// ---------------------------------------------------------------------------
// Fragment assembly, same contiguous k order for A and B (wave32, 16-bit A 16x32 / B 32x16):
//  A: lane L (M = L%16): elems 0..7 = K[kb + 16*(L>=16) .. +7],
//                        elems 8..15 = K[kb + 16*(L>=16) + 8 .. +7]
//  B (held as Bt[N][K]): lane L (N = L%16): elems 0..15 = K[kb + 16*(L>=16) ..]
// Both reduce to two aligned 16-byte loads per fragment.
// ---------------------------------------------------------------------------
__device__ __forceinline__ v16bf make_frag(const __bf16* p0, const __bf16* p1) {
  v8bf lo = *(const v8bf*)p0;
  v8bf hi = *(const v8bf*)p1;
  return __builtin_shufflevector(lo, hi, 0,1,2,3,4,5,6,7,8,9,10,11,12,13,14,15);
}

// C = A[M][K] @ Bt[N][K]^T + bias[N]  -> C[M][N] fp32. grid=(N/128, M/16), block=256 (8 waves)
__global__ void wmma_gemm_bias(const __bf16* __restrict__ A, const __bf16* __restrict__ Bt,
                               const float* __restrict__ bias, float* __restrict__ C,
                               int N, int K) {
  const int lane  = threadIdx.x & 31;
  const int wv    = threadIdx.x >> 5;
  const int ntile = blockIdx.x * 8 + wv;
  const int mtile = blockIdx.y;
  const int nl = lane & 15, hb = lane >> 4;
  const __bf16* Arow = A  + (size_t)(mtile * 16 + nl) * K;
  const __bf16* Brow = Bt + (size_t)(ntile * 16 + nl) * K;
  v8f acc = {};
  for (int k0 = 0; k0 < K; k0 += 32) {
    v16bf a = make_frag(Arow + k0 + 16 * hb, Arow + k0 + 16 * hb + 8);
    v16bf b = make_frag(Brow + k0 + 16 * hb, Brow + k0 + 16 * hb + 8);
    acc = __builtin_amdgcn_wmma_f32_16x16x32_bf16(false, a, false, b, (short)0, acc, false, false);
  }
  const int col = ntile * 16 + nl;
  const float bv = bias[col];
#pragma unroll
  for (int i = 0; i < 8; ++i) {
    int row = mtile * 16 + 8 * hb + i;
    C[(size_t)row * N + col] = acc[i] + bv;
  }
}

// One GRU step (keras reset_after=true). Computes MI = h@U for all three gate
// blocks with WMMA, fuses gate math, writes new state (fp32 + bf16, ping/pong).
// Encoder mode: MX != null (precomputed x@W+b0, rows b*TT+t), writes hpe[b][t][:].
// Decoder mode: Eall != null (precomputed emb@W+b0 per vocab row), gathers via idx.
// grid=(8,4) block=256: 64 n-tiles x 4 m-tiles.
__global__ void gru_step_kernel(const __bf16* __restrict__ hA, const float* __restrict__ hPrev,
                                const __bf16* __restrict__ Ut, const float* __restrict__ b1,
                                const float* __restrict__ MX, const float* __restrict__ Eall,
                                const int* __restrict__ idx, int t,
                                float* __restrict__ hNext, __bf16* __restrict__ hNextB,
                                float* __restrict__ hpe) {
  const int lane  = threadIdx.x & 31;
  const int wv    = threadIdx.x >> 5;
  const int ntile = blockIdx.x * 8 + wv;   // 0..63
  const int mtile = blockIdx.y;            // 0..3
  const int nl = lane & 15, hb = lane >> 4;
  const __bf16* Arow = hA + (size_t)(mtile * 16 + nl) * HH;
  const __bf16* Bz = Ut + (size_t)(ntile * 16 + nl) * HH;
  const __bf16* Br = Bz + (size_t)HH * HH;
  const __bf16* Bh = Br + (size_t)HH * HH;
  v8f az = {}, ar = {}, ah = {};
  for (int k0 = 0; k0 < HH; k0 += 32) {
    v16bf a  = make_frag(Arow + k0 + 16 * hb, Arow + k0 + 16 * hb + 8);
    v16bf bz = make_frag(Bz + k0 + 16 * hb, Bz + k0 + 16 * hb + 8);
    v16bf br = make_frag(Br + k0 + 16 * hb, Br + k0 + 16 * hb + 8);
    v16bf bh = make_frag(Bh + k0 + 16 * hb, Bh + k0 + 16 * hb + 8);
    az = __builtin_amdgcn_wmma_f32_16x16x32_bf16(false, a, false, bz, (short)0, az, false, false);
    ar = __builtin_amdgcn_wmma_f32_16x16x32_bf16(false, a, false, br, (short)0, ar, false, false);
    ah = __builtin_amdgcn_wmma_f32_16x16x32_bf16(false, a, false, bh, (short)0, ah, false, false);
  }
  const int col = ntile * 16 + nl;
  const float b1z = b1[col], b1r = b1[HH + col], b1h = b1[2 * HH + col];
#pragma unroll
  for (int i = 0; i < 8; ++i) {
    const int row = mtile * 16 + 8 * hb + i;   // batch index 0..63
    float mxz, mxr, mxh;
    if (MX) {
      const float* m = MX + (size_t)(row * TT + t) * H3;
      mxz = m[col]; mxr = m[HH + col]; mxh = m[2 * HH + col];
    } else {
      const float* m = Eall + (size_t)idx[row] * H3;
      mxz = m[col]; mxr = m[HH + col]; mxh = m[2 * HH + col];
    }
    const float z  = 1.f / (1.f + expf(-(mxz + az[i] + b1z)));
    const float r  = 1.f / (1.f + expf(-(mxr + ar[i] + b1r)));
    const float hh = tanhf(mxh + r * (ah[i] + b1h));
    const float hp = hPrev[(size_t)row * HH + col];
    const float hn = z * hp + (1.f - z) * hh;
    hNext[(size_t)row * HH + col]  = hn;
    hNextB[(size_t)row * HH + col] = (__bf16)hn;
    if (hpe) hpe[((size_t)row * TT + t) * HH + col] = hn;
  }
}

// ------------------------- small prep / glue kernels -----------------------
__global__ void f32_to_bf16_kernel(const float* __restrict__ in, __bf16* __restrict__ out, int n) {
  int i = blockIdx.x * 256 + threadIdx.x;
  if (i < n) out[i] = (__bf16)in[i];
}

// in [R][C] fp32 row-major -> out [C][R] bf16
__global__ void transpose_to_bf16_kernel(const float* __restrict__ in, __bf16* __restrict__ out,
                                         int R, int C) {
  int i = blockIdx.x * 256 + threadIdx.x;
  if (i >= R * C) return;
  int r = i / C, c = i - r * C;
  out[(size_t)c * R + r] = (__bf16)in[i];
}

__global__ void emb_pad_kernel(const float* __restrict__ emb, __bf16* __restrict__ out) {
  int i = blockIdx.x * 256 + threadIdx.x;   // 32*1024
  if (i >= 32 * HH) return;
  int v = i >> 10, k = i & (HH - 1);
  out[i] = (v < OUTD) ? (__bf16)emb[v * HH + k] : (__bf16)0.0f;
}

// scores[b][j] = sum_d scale[d]*tanh(hd[b][d] + hpe[b][j][d]); grid=(TT,BB), block=256
__global__ void attn_scores_kernel(const float* __restrict__ hd, const float* __restrict__ hpe,
                                   const float* __restrict__ scale, float* __restrict__ scores) {
  const int j = blockIdx.x, b = blockIdx.y;
  __shared__ float red[256];
  const float* hp = hpe + ((size_t)b * TT + j) * HH;
  const float* hb = hd + (size_t)b * HH;
  float p = 0.f;
  for (int d = threadIdx.x; d < HH; d += 256) p += scale[d] * tanhf(hb[d] + hp[d]);
  red[threadIdx.x] = p; __syncthreads();
  for (int s = 128; s > 0; s >>= 1) {
    if (threadIdx.x < s) red[threadIdx.x] += red[threadIdx.x + s];
    __syncthreads();
  }
  if (threadIdx.x == 0) scores[b * TT + j] = red[0];
}

__global__ void softmax_kernel(const float* __restrict__ scores, float* __restrict__ attn) {
  const int b = blockIdx.x, j = threadIdx.x;
  __shared__ float red[128];
  float v = (j < TT) ? scores[b * TT + j] : -3.0e38f;
  red[j] = v; __syncthreads();
  for (int s = 64; s > 0; s >>= 1) { if (j < s) red[j] = fmaxf(red[j], red[j + s]); __syncthreads(); }
  float m = red[0]; __syncthreads();
  float e = (j < TT) ? expf(v - m) : 0.f;
  red[j] = e; __syncthreads();
  for (int s = 64; s > 0; s >>= 1) { if (j < s) red[j] += red[j + s]; __syncthreads(); }
  if (j < TT) attn[b * TT + j] = e / red[0];
}

__global__ void ctx_kernel(const float* __restrict__ attn, const float* __restrict__ hpe,
                           float* __restrict__ ctx) {
  int g = blockIdx.x * 256 + threadIdx.x;  // 64*1024
  if (g >= BB * HH) return;
  int b = g >> 10, d = g & (HH - 1);
  const float* a  = attn + b * TT;
  const float* hp = hpe + (size_t)b * TT * HH + d;
  float s = 0.f;
  for (int j = 0; j < TT; ++j) s += a[j] * hp[(size_t)j * HH];
  ctx[g] = s;
}

__global__ void fc_kernel(const float* __restrict__ hd, const float* __restrict__ ctx,
                          const float* __restrict__ W, const float* __restrict__ bias,
                          float* __restrict__ out, float* __restrict__ pbuf, int t) {
  const int b = blockIdx.x, o = threadIdx.x;
  if (o >= OUTD) return;
  float acc = bias[o];
  const float* hb = hd + (size_t)b * HH;
  const float* cb = ctx + (size_t)b * HH;
  for (int k = 0; k < HH; ++k) acc += hb[k] * W[(size_t)k * OUTD + o];
  for (int k = 0; k < HH; ++k) acc += cb[k] * W[(size_t)(HH + k) * OUTD + o];
  out[((size_t)b * TT + t) * OUTD + o] = acc;
  pbuf[b * OUTD + o] = acc;
}

__global__ void argmax_kernel(const float* __restrict__ p, int* __restrict__ idx) {
  int b = blockIdx.x * 32 + threadIdx.x;
  if (b >= BB) return;
  const float* pb = p + b * OUTD;
  float best = pb[0]; int bi = 0;
  for (int o = 1; o < OUTD; ++o) if (pb[o] > best) { best = pb[o]; bi = o; }
  idx[b] = bi;
}

// ---------------------------------------------------------------------------
extern "C" void kernel_launch(void* const* d_in, const int* in_sizes, int n_in,
                              void* d_out, int out_size, void* d_ws, size_t ws_size,
                              hipStream_t stream) {
  const float* x      = (const float*)d_in[0];
  const float* encW   = (const float*)d_in[1];
  const float* encU   = (const float*)d_in[2];
  const float* encb   = (const float*)d_in[3];   // [2][3H]
  const float* ascale = (const float*)d_in[4];
  const float* embW   = (const float*)d_in[5];
  const float* decW   = (const float*)d_in[6];
  const float* decU   = (const float*)d_in[7];
  const float* decb   = (const float*)d_in[8];   // [2][3H]
  const float* fcW    = (const float*)d_in[9];
  const float* fcb    = (const float*)d_in[10];
  float* out = (float*)d_out;
  (void)in_sizes; (void)n_in; (void)out_size; (void)ws_size;

  char* base = (char*)d_ws; size_t off = 0;
  auto alloc = [&](size_t bytes) -> void* {
    void* p = base + off; off = (off + bytes + 255) & ~(size_t)255; return p;
  };
  __bf16* xb    = (__bf16*)alloc((size_t)BB * TT * DD * 2);
  __bf16* encWt = (__bf16*)alloc((size_t)H3 * DD * 2);
  __bf16* encUt = (__bf16*)alloc((size_t)H3 * HH * 2);
  __bf16* decWt = (__bf16*)alloc((size_t)H3 * HH * 2);
  __bf16* decUt = (__bf16*)alloc((size_t)H3 * HH * 2);
  __bf16* embP  = (__bf16*)alloc((size_t)32 * HH * 2);
  float*  MX    = (float*)alloc((size_t)BB * TT * H3 * 4);
  float*  Eall  = (float*)alloc((size_t)32 * H3 * 4);
  float*  hpe   = (float*)alloc((size_t)BB * TT * HH * 4);
  float*  hF[2]; __bf16* hB[2]; float* dF[2]; __bf16* dB[2];
  for (int i = 0; i < 2; ++i) { hF[i] = (float*)alloc((size_t)BB * HH * 4); hB[i] = (__bf16*)alloc((size_t)BB * HH * 2); }
  for (int i = 0; i < 2; ++i) { dF[i] = (float*)alloc((size_t)BB * HH * 4); dB[i] = (__bf16*)alloc((size_t)BB * HH * 2); }
  float* scores = (float*)alloc((size_t)BB * TT * 4);
  float* attn   = (float*)alloc((size_t)BB * TT * 4);
  float* ctx    = (float*)alloc((size_t)BB * HH * 4);
  float* pbuf   = (float*)alloc((size_t)BB * OUTD * 4);
  int*   idx    = (int*)alloc((size_t)BB * 4);

  // ---- one-time prep (recomputed every call: deterministic, capture-safe) ----
  f32_to_bf16_kernel<<<(BB * TT * DD + 255) / 256, 256, 0, stream>>>(x, xb, BB * TT * DD);
  transpose_to_bf16_kernel<<<(DD * H3 + 255) / 256, 256, 0, stream>>>(encW, encWt, DD, H3);
  transpose_to_bf16_kernel<<<(HH * H3 + 255) / 256, 256, 0, stream>>>(encU, encUt, HH, H3);
  transpose_to_bf16_kernel<<<(HH * H3 + 255) / 256, 256, 0, stream>>>(decW, decWt, HH, H3);
  transpose_to_bf16_kernel<<<(HH * H3 + 255) / 256, 256, 0, stream>>>(decU, decUt, HH, H3);
  emb_pad_kernel<<<(32 * HH + 255) / 256, 256, 0, stream>>>(embW, embP);
  hipMemsetAsync(hF[0], 0, (size_t)BB * HH * 4, stream);
  hipMemsetAsync(hB[0], 0, (size_t)BB * HH * 2, stream);
  hipMemsetAsync(dF[0], 0, (size_t)BB * HH * 4, stream);
  hipMemsetAsync(dB[0], 0, (size_t)BB * HH * 2, stream);
  hipMemsetAsync(idx, 0, (size_t)BB * 4, stream);   // argmax(zeros) == 0

  // MX = x @ enc_W + enc_b[0] for all (b,t): one parallel 6400x3072x1024 WMMA GEMM
  wmma_gemm_bias<<<dim3(H3 / 128, (BB * TT) / 16), 256, 0, stream>>>(xb, encWt, encb, MX, H3, DD);
  // E_all = emb_W @ dec_W + dec_b[0] for all vocab rows (decoder input GEMM hoisted)
  wmma_gemm_bias<<<dim3(H3 / 128, 2), 256, 0, stream>>>(embP, decWt, decb, Eall, H3, HH);

  // ---- encoder recurrence ----
  int cur = 0;
  for (int t = 0; t < TT; ++t) {
    gru_step_kernel<<<dim3(8, 4), 256, 0, stream>>>(
        hB[cur], hF[cur], encUt, encb + H3, MX, nullptr, nullptr, t,
        hF[1 - cur], hB[1 - cur], hpe);
    cur ^= 1;
  }

  // ---- decoder recurrence ----
  int dc = 0;
  for (int t = 0; t < TT; ++t) {
    attn_scores_kernel<<<dim3(TT, BB), 256, 0, stream>>>(dF[dc], hpe, ascale, scores);
    softmax_kernel<<<BB, 128, 0, stream>>>(scores, attn);
    ctx_kernel<<<(BB * HH + 255) / 256, 256, 0, stream>>>(attn, hpe, ctx);
    gru_step_kernel<<<dim3(8, 4), 256, 0, stream>>>(
        dB[dc], dF[dc], decUt, decb + H3, nullptr, Eall, idx, t,
        dF[1 - dc], dB[1 - dc], nullptr);
    fc_kernel<<<BB, 32, 0, stream>>>(dF[1 - dc], ctx, fcW, fcb, out, pbuf, t);
    argmax_kernel<<<(BB + 31) / 32, 32, 0, stream>>>(pbuf, idx);
    dc ^= 1;
  }
}
